// AxialSelfAttentionModule_2482491097685
// MI455X (gfx1250) — matrix operand order repaired and checked by reading, compiled, not run on hardware
//
#include <hip/hip_runtime.h>
#include <hip/hip_bf16.h>

// ---------------------------------------------------------------------------
// Types for CDNA5 WMMA
// ---------------------------------------------------------------------------
typedef __bf16 bf16_t;
typedef bf16_t v16bf __attribute__((ext_vector_type(16)));   // 32 B, one A/B fragment per lane
typedef float  v8f   __attribute__((ext_vector_type(8)));    // 32 B, C/D fragment per lane
typedef unsigned short v8us __attribute__((ext_vector_type(8))); // 16 B packed bf16 store

#define CC    192
#define VV    32768          // 32^3
#define BB    2
#define NHH   8
#define HDD   24
#define QKV_M 576

// ---------------------------------------------------------------------------
// bf16 helpers (bit-level, RNE)
// ---------------------------------------------------------------------------
__device__ __forceinline__ unsigned short f2bf(float f) {
    unsigned int u = __float_as_uint(f);
    u += 0x7FFFu + ((u >> 16) & 1u);
    return (unsigned short)(u >> 16);
}
__device__ __forceinline__ float bf2f(unsigned short u) {
    return __uint_as_float(((unsigned int)u) << 16);
}
__device__ __forceinline__ float gelu_exact(float x) {
    return 0.5f * x * (1.0f + erff(x * 0.70710678118654752f));
}

// ---------------------------------------------------------------------------
// Elementwise utilities
// ---------------------------------------------------------------------------
__global__ void k_cvt_f32_bf16(const float* __restrict__ in,
                               unsigned short* __restrict__ out, int n) {
    int i = blockIdx.x * blockDim.x + threadIdx.x;
    if (i < n) out[i] = f2bf(in[i]);
}
__global__ void k_zero_f32(float* __restrict__ p, int n) {
    int i = blockIdx.x * blockDim.x + threadIdx.x;
    if (i < n) p[i] = 0.0f;
}

// ---------------------------------------------------------------------------
// Channel-major (B,C,V) f32 -> spatial-major (B,V,C) bf16, LDS 32x32 tiles
// grid = (V/32, C/32, B), block = 256 (8x32)
// ---------------------------------------------------------------------------
__global__ void k_tr_cvt(const float* __restrict__ in, unsigned short* __restrict__ out) {
    __shared__ float tile[32][33];
    const int tx = threadIdx.x & 31, ty = threadIdx.x >> 5;
    const int s0 = blockIdx.x * 32, c0 = blockIdx.y * 32, b = blockIdx.z;
#pragma unroll
    for (int r = 0; r < 4; ++r)
        tile[ty + r * 8][tx] = in[((size_t)b * CC + c0 + ty + r * 8) * VV + s0 + tx];
    __syncthreads();
#pragma unroll
    for (int r = 0; r < 4; ++r)
        out[((size_t)b * VV + s0 + ty + r * 8) * CC + c0 + tx] = f2bf(tile[tx][ty + r * 8]);
}

// ---------------------------------------------------------------------------
// Weight swizzle: fp32 (M,K) row-major -> bf16 WMMA A-fragment order
//   frag[((mt*KC + kc)*32 + lane)*16 + e] = W[(mt*16 + lane&15)*K + kc*32 + Kperm]
//   Kperm = e + 16*(lane>>4)   (contiguous k order, matches B loads)
// ---------------------------------------------------------------------------
__global__ void k_swz_w(const float* __restrict__ W, unsigned short* __restrict__ F,
                        int K, int n) {
    int idx = blockIdx.x * blockDim.x + threadIdx.x;
    if (idx >= n) return;
    const int KC = K >> 5;
    int e = idx & 15, l = (idx >> 4) & 31, rest = idx >> 9;
    int kc = rest % KC, mt = rest / KC;
    int row = mt * 16 + (l & 15);
    int kk  = kc * 32 + e + 16 * (l >> 4);
    F[idx] = f2bf(W[(size_t)row * K + kk]);
}
// lp1_w (C,C,3,3,3): K-space = 27 taps x (192/32) chunks, channel-contiguous frags
__global__ void k_swz_lp1(const float* __restrict__ W, unsigned short* __restrict__ F, int n) {
    int idx = blockIdx.x * blockDim.x + threadIdx.x;
    if (idx >= n) return;
    const int KC = 27 * 6;
    int e = idx & 15, l = (idx >> 4) & 31, rest = idx >> 9;
    int kc = rest % KC, mt = rest / KC;
    int t  = kc / 6;
    int ch = (kc % 6) * 32 + e + 16 * (l >> 4);
    int row = mt * 16 + (l & 15);
    F[idx] = f2bf(W[((size_t)row * CC + ch) * 27 + t]);
}

// ---------------------------------------------------------------------------
// InstanceNorm stats, spatial-major input (B,V,C) f32.
// Phase 1: block = 192 threads (one per channel), coalesced; direct partial
//          stores (no float atomics -> deterministic). grid = (V/128, B)
// Phase 2: 256-way tree reduce -> (mean, rsqrt) per (b,c). grid = B*C
// Phase 3: coalesced normalize + GELU -> bf16
// ---------------------------------------------------------------------------
__global__ void k_stat_part(const float* __restrict__ in, float2* __restrict__ part) {
    const int c = threadIdx.x, chunk = blockIdx.x, b = blockIdx.y;
    const int s0 = chunk * 128;
    float sum = 0.f, ss = 0.f;
    for (int s = s0; s < s0 + 128; ++s) {
        float v = in[((size_t)b * VV + s) * CC + c];
        sum += v; ss += v * v;
    }
    part[((size_t)b * CC + c) * 256 + chunk] = make_float2(sum, ss);
}
__global__ void k_stat_reduce(const float2* __restrict__ part, float2* __restrict__ stats) {
    const int bc = blockIdx.x;
    __shared__ float s1[256], s2[256];
    float2 p = part[(size_t)bc * 256 + threadIdx.x];
    s1[threadIdx.x] = p.x; s2[threadIdx.x] = p.y;
    __syncthreads();
    for (int st = 128; st > 0; st >>= 1) {
        if (threadIdx.x < st) {
            s1[threadIdx.x] += s1[threadIdx.x + st];
            s2[threadIdx.x] += s2[threadIdx.x + st];
        }
        __syncthreads();
    }
    if (threadIdx.x == 0) {
        float mean = s1[0] * (1.0f / VV);
        float var  = s2[0] * (1.0f / VV) - mean * mean;
        stats[bc] = make_float2(mean, rsqrtf(var + 1e-5f));
    }
}
__global__ void k_norm_gelu(const float* __restrict__ in, const float2* __restrict__ stats,
                            unsigned short* __restrict__ out, int n) {
    int i = blockIdx.x * blockDim.x + threadIdx.x;
    if (i >= n) return;
    int c = i % CC;
    int b = i / (VV * CC);
    float2 st = stats[b * CC + c];
    float v = (in[i] - st.x) * st.y;
    out[i] = f2bf(gelu_exact(v));
}

// ---------------------------------------------------------------------------
// Dense bf16 WMMA GEMM:  acc[o, s] = sum_k W[o,k]*X[b,s,k] + bias[o]
//   Wf: pre-swizzled bf16 fragments; X: (B,V,K) bf16 spatial-major.
//   One wave = one 16x16 tile; inner loop = 2 wide loads + 1 v_wmma.
//   grid = (V/64, M/16, B), block = 128 (4 waves -> 4 adjacent N tiles)
// EPI: 0 = f32 spatial-major (v8f store)
//      1 = bf16 spatial-major (v8us store)
//      2 = bf16 spatial-major of xmul * sigmoid(acc)   (xmul channel-major)
//      3 = f32 channel-major (final output)
// ---------------------------------------------------------------------------
template <int EPI>
__global__ void k_gemm_bf16(const unsigned short* __restrict__ Wf,
                            const float* __restrict__ bias,
                            const unsigned short* __restrict__ X,
                            void* __restrict__ Out,
                            const float* __restrict__ xmul,
                            int M, int K) {
    const int lane   = threadIdx.x & 31;
    const int wid    = threadIdx.x >> 5;
    const int laneHi = lane >> 4;
    const int lane15 = lane & 15;
    const int mt = blockIdx.y;
    const int m0 = mt * 16;
    const int n0 = (blockIdx.x * 4 + wid) * 16;
    const int b  = blockIdx.z;
    const int KC = K >> 5;

    const unsigned short* Xb = X + ((size_t)b * VV + n0 + lane15) * K + 16 * laneHi;
    const unsigned short* Wp = Wf + (size_t)mt * KC * 512 + lane * 16;

    v8f acc;
#pragma unroll
    for (int r = 0; r < 8; ++r) acc[r] = bias[m0 + r + 8 * laneHi];

    for (int kc = 0; kc < KC; ++kc) {
        v16bf a  = *reinterpret_cast<const v16bf*>(Wp + (size_t)kc * 512);
        v16bf bm = *reinterpret_cast<const v16bf*>(Xb + kc * 32);
        acc = __builtin_amdgcn_wmma_f32_16x16x32_bf16(
            false, a, false, bm, (short)0, acc, false, false);
    }

    const int col  = n0 + lane15;
    const int row0 = m0 + 8 * laneHi;
    if constexpr (EPI == 0) {
        *reinterpret_cast<v8f*>(&((float*)Out)[((size_t)b * VV + col) * M + row0]) = acc;
    } else if constexpr (EPI == 1) {
        v8us pk;
#pragma unroll
        for (int r = 0; r < 8; ++r) pk[r] = f2bf(acc[r]);
        *reinterpret_cast<v8us*>(&((unsigned short*)Out)[((size_t)b * VV + col) * M + row0]) = pk;
    } else if constexpr (EPI == 2) {
        v8us pk;
#pragma unroll
        for (int r = 0; r < 8; ++r) {
            float s = 1.0f / (1.0f + __expf(-acc[r]));
            pk[r] = f2bf(xmul[((size_t)b * M + row0 + r) * VV + col] * s);
        }
        *reinterpret_cast<v8us*>(&((unsigned short*)Out)[((size_t)b * VV + col) * M + row0]) = pk;
    } else {
#pragma unroll
        for (int r = 0; r < 8; ++r)
            ((float*)Out)[((size_t)b * M + row0 + r) * VV + col] = acc[r];
    }
}

// ---------------------------------------------------------------------------
// Circular 3x3x3 conv as implicit WMMA GEMM (K = 27 taps x 192 channels)
//   Wt: swizzled (12, 162, 32, 16) bf16 fragments; X: (B,V,192) bf16
//   Out: (B,V,192) f32 spatial-major.  grid = (V/64, 12, B), block = 128
// ---------------------------------------------------------------------------
__global__ void k_conv3_wmma(const unsigned short* __restrict__ Wt,
                             const float* __restrict__ bias,
                             const unsigned short* __restrict__ X,
                             float* __restrict__ Out) {
    const int lane   = threadIdx.x & 31;
    const int wid    = threadIdx.x >> 5;
    const int laneHi = lane >> 4;
    const int lane15 = lane & 15;
    const int m0 = blockIdx.y * 16;
    const int n0 = (blockIdx.x * 4 + wid) * 16;
    const int b  = blockIdx.z;

    const int n = n0 + lane15;
    const int d = n >> 10, h = (n >> 5) & 31, w = n & 31;

    const unsigned short* Wp = Wt + (size_t)blockIdx.y * 162 * 512 + lane * 16;

    v8f acc;
#pragma unroll
    for (int r = 0; r < 8; ++r) acc[r] = bias[m0 + r + 8 * laneHi];

    for (int t = 0; t < 27; ++t) {
        const int kd = t / 9, kh = (t % 9) / 3, kw = t % 3;
        const int dd = (d + kd - 1 + 32) & 31;
        const int hh = (h + kh - 1 + 32) & 31;
        const int ww = (w + kw - 1 + 32) & 31;
        const int vox = dd * 1024 + hh * 32 + ww;
        const unsigned short* Xb = X + ((size_t)b * VV + vox) * CC + 16 * laneHi;
#pragma unroll
        for (int kc = 0; kc < 6; ++kc) {
            v16bf a  = *reinterpret_cast<const v16bf*>(Wp + (size_t)(t * 6 + kc) * 512);
            v16bf bm = *reinterpret_cast<const v16bf*>(Xb + kc * 32);
            acc = __builtin_amdgcn_wmma_f32_16x16x32_bf16(
                false, a, false, bm, (short)0, acc, false, false);
        }
    }
    *reinterpret_cast<v8f*>(
        &Out[((size_t)b * VV + n0 + lane15) * CC + m0 + 8 * laneHi]) = acc;
}

// ---------------------------------------------------------------------------
// Axial attention with LIERE rotation (one wave = one line/head).
//   qkv: bf16 (B, V, 576) spatial-major -> q/k/v are 3x16B contiguous loads.
//   The reference's "+pa" bias is constant along the softmax axis -> dropped.
// ---------------------------------------------------------------------------
__device__ __forceinline__ void load24bf(const unsigned short* p, float* f) {
    const uint4* p4 = reinterpret_cast<const uint4*>(p);
#pragma unroll
    for (int j = 0; j < 3; ++j) {
        uint4 u = p4[j];
        f[j * 8 + 0] = bf2f((unsigned short)u.x);  f[j * 8 + 1] = bf2f((unsigned short)(u.x >> 16));
        f[j * 8 + 2] = bf2f((unsigned short)u.y);  f[j * 8 + 3] = bf2f((unsigned short)(u.y >> 16));
        f[j * 8 + 4] = bf2f((unsigned short)u.z);  f[j * 8 + 5] = bf2f((unsigned short)(u.z >> 16));
        f[j * 8 + 6] = bf2f((unsigned short)u.w);  f[j * 8 + 7] = bf2f((unsigned short)(u.w >> 16));
    }
}

__global__ void k_attn_axis(const unsigned short* __restrict__ qkv,
                            const float* __restrict__ A,
                            int mode, int comp,
                            float* __restrict__ outAcc) {      // (B,V,192) f32, +=
    const int l   = threadIdx.x;
    const int lid = blockIdx.x;
    const int hh  = blockIdx.y;
    const int b   = blockIdx.z;

    int vox;
    if (mode == 0)      vox = l * 1024 + lid;
    else if (mode == 1) vox = (lid >> 5) * 1024 + l * 32 + (lid & 31);
    else                vox = lid * 32 + l;

    const float p  = -1.0f + 2.0f * (float)l / 31.0f;
    const float ax = A[hh * 9 + 0 * 3 + comp];
    const float ay = A[hh * 9 + 1 * 3 + comp];
    const float az = A[hh * 9 + 2 * 3 + comp];
    float wx = p * ax, wy = p * ay, wz = p * az;
    float th = fmaxf(sqrtf(wx * wx + wy * wy + wz * wz), 1e-8f);
    float ux = wx / th, uy = wy / th, uz = wz / th;
    float sn = sinf(th), cs = cosf(th), d1 = 1.0f - cs;
    float R00 = cs + d1 * ux * ux,      R01 = -sn * uz + d1 * ux * uy, R02 =  sn * uy + d1 * ux * uz;
    float R10 =  sn * uz + d1 * uy * ux, R11 = cs + d1 * uy * uy,      R12 = -sn * ux + d1 * uy * uz;
    float R20 = -sn * uy + d1 * uz * ux, R21 =  sn * ux + d1 * uz * uy, R22 = cs + d1 * uz * uz;

    const unsigned short* base = qkv + ((size_t)b * VV + vox) * QKV_M + hh * HDD;

    __shared__ float kr_s[32 * HDD];
    __shared__ float vv_s[32 * HDD];

    float qv[HDD], kv[HDD], vv[HDD], qr[HDD];
    load24bf(base,           qv);
    load24bf(base + CC,      kv);
    load24bf(base + 2 * CC,  vv);
#pragma unroll
    for (int g = 0; g < HDD / 3; ++g) {
        float x0 = qv[3 * g], x1 = qv[3 * g + 1], x2 = qv[3 * g + 2];
        qr[3 * g + 0] = R00 * x0 + R01 * x1 + R02 * x2;
        qr[3 * g + 1] = R10 * x0 + R11 * x1 + R12 * x2;
        qr[3 * g + 2] = R20 * x0 + R21 * x1 + R22 * x2;
        float y0 = kv[3 * g], y1 = kv[3 * g + 1], y2 = kv[3 * g + 2];
        kr_s[l * HDD + 3 * g + 0] = R00 * y0 + R01 * y1 + R02 * y2;
        kr_s[l * HDD + 3 * g + 1] = R10 * y0 + R11 * y1 + R12 * y2;
        kr_s[l * HDD + 3 * g + 2] = R20 * y0 + R21 * y1 + R22 * y2;
        vv_s[l * HDD + 3 * g + 0] = vv[3 * g];
        vv_s[l * HDD + 3 * g + 1] = vv[3 * g + 1];
        vv_s[l * HDD + 3 * g + 2] = vv[3 * g + 2];
    }
    __syncthreads();

    const float scale = 0.20412414523193154f;   // 24^-0.5
    float sc[32];
    float mx = -1e30f;
#pragma unroll
    for (int m = 0; m < 32; ++m) {
        float dot = 0.f;
#pragma unroll
        for (int d2 = 0; d2 < HDD; ++d2) dot += qr[d2] * kr_s[m * HDD + d2];
        sc[m] = dot * scale;
        mx = fmaxf(mx, sc[m]);
    }
    float se = 0.f;
#pragma unroll
    for (int m = 0; m < 32; ++m) { sc[m] = __expf(sc[m] - mx); se += sc[m]; }
    const float inv = 1.0f / se;

    float o[HDD];
#pragma unroll
    for (int d2 = 0; d2 < HDD; ++d2) o[d2] = 0.f;
#pragma unroll
    for (int m = 0; m < 32; ++m) {
        float wgt = sc[m] * inv;
#pragma unroll
        for (int d2 = 0; d2 < HDD; ++d2) o[d2] += wgt * vv_s[m * HDD + d2];
    }
    float* ob = outAcc + ((size_t)b * VV + vox) * CC + hh * HDD;
#pragma unroll
    for (int d2 = 0; d2 < HDD; ++d2) ob[d2] += o[d2];
}

// ---------------------------------------------------------------------------
// Host-side orchestration
// ---------------------------------------------------------------------------
extern "C" void kernel_launch(void* const* d_in, const int* in_sizes, int n_in,
                              void* d_out, int out_size, void* d_ws, size_t ws_size,
                              hipStream_t stream) {
    const float* x      = (const float*)d_in[0];
    const float* pos    = (const float*)d_in[1];
    const float* lp1_w  = (const float*)d_in[2];
    const float* lp1_b  = (const float*)d_in[3];
    const float* lp2_w  = (const float*)d_in[4];
    const float* lp2_b  = (const float*)d_in[5];
    const float* m1_w   = (const float*)d_in[6];
    const float* m1_b   = (const float*)d_in[7];
    const float* m2_w   = (const float*)d_in[8];
    const float* m2_b   = (const float*)d_in[9];
    // d_in[10]=pa_w, d_in[11]=pa_b : bias is constant along softmax axis -> no-op
    const float* qkv_w  = (const float*)d_in[12];
    const float* qkv_b  = (const float*)d_in[13];
    const float* A_d    = (const float*)d_in[14];
    const float* A_h    = (const float*)d_in[15];
    const float* A_w    = (const float*)d_in[16];
    const float* proj_w = (const float*)d_in[17];
    const float* proj_b = (const float*)d_in[18];

    const int NTOT = BB * CC * VV;               // 12,582,912
    char* ws = (char*)d_ws;
    float*          bufA   = (float*)(ws + 0);                     // 50,331,648 B (f32 act / attn accum)
    unsigned short* bufH   = (unsigned short*)(ws + 50331648);     // 25,165,824 B (bf16 act)
    unsigned short* bufX   = (unsigned short*)(ws + 75497472);     // 25,165,824 B (x_mod bf16)
    unsigned short* bufQ   = (unsigned short*)(ws + 100663296);    // 75,497,472 B (qkv bf16)
    unsigned short* wfLp2  = (unsigned short*)(ws + 176160768);    //     73,728 B
    unsigned short* wfM1   = (unsigned short*)(ws + 176234496);
    unsigned short* wfM2   = (unsigned short*)(ws + 176308224);
    unsigned short* wfProj = (unsigned short*)(ws + 176381952);
    unsigned short* wfQkv  = (unsigned short*)(ws + 176455680);    //    221,184 B
    unsigned short* wfLp1  = (unsigned short*)(ws + 176676864);    //  1,990,656 B
    float2*         part   = (float2*)(ws + 178667520);            //    786,432 B
    float2*         stats  = (float2*)(ws + 179453952);            //      3,072 B

    const dim3 blk128(128), blk256(256), blk192(192), blk32(32);
    const dim3 gN((NTOT + 255) / 256);
    const dim3 gGemm192(VV / 64, CC / 16, BB);
    const dim3 gGemmQKV(VV / 64, QKV_M / 16, BB);
    const dim3 gTr(VV / 32, CC / 32, BB);
    const dim3 gStat(VV / 128, BB);
    const dim3 gRed(BB * CC);
    const dim3 gAttn(1024, NHH, BB);

    // --- weight prep: bf16 + WMMA-fragment swizzle (tiny, once per call) ---
    const int nW192 = CC * CC, nWq = QKV_M * CC, nW1 = 12 * 162 * 512;
    k_swz_w  <<<dim3((nW192 + 255) / 256), blk256, 0, stream>>>(lp2_w,  wfLp2,  CC, nW192);
    k_swz_w  <<<dim3((nW192 + 255) / 256), blk256, 0, stream>>>(m1_w,   wfM1,   CC, nW192);
    k_swz_w  <<<dim3((nW192 + 255) / 256), blk256, 0, stream>>>(m2_w,   wfM2,   CC, nW192);
    k_swz_w  <<<dim3((nW192 + 255) / 256), blk256, 0, stream>>>(proj_w, wfProj, CC, nW192);
    k_swz_w  <<<dim3((nWq  + 255) / 256), blk256, 0, stream>>>(qkv_w,  wfQkv,  CC, nWq);
    k_swz_lp1<<<dim3((nW1  + 255) / 256), blk256, 0, stream>>>(lp1_w,  wfLp1,  nW1);

    // 0) pos_emb (B,C,V) f32 -> (B,V,C) bf16
    k_tr_cvt<<<gTr, blk256, 0, stream>>>(pos, bufH);
    // 1) t1 = conv3_circ(pos_emb)   [WMMA implicit GEMM, 162 chunks]
    k_conv3_wmma<<<gGemm192, blk128, 0, stream>>>(wfLp1, lp1_b, bufH, bufA);
    // 2) gelu(inorm(t1)) -> bf16
    k_stat_part  <<<gStat, blk192, 0, stream>>>(bufA, part);
    k_stat_reduce<<<gRed,  blk256, 0, stream>>>(part, stats);
    k_norm_gelu  <<<gN,    blk256, 0, stream>>>(bufA, stats, bufH, NTOT);
    // 3) local = lp2 GEMM -> f32
    k_gemm_bf16<0><<<gGemm192, blk128, 0, stream>>>(wfLp2, lp2_b, bufH, bufA, nullptr, CC, CC);
    // 4) local -> bf16
    k_cvt_f32_bf16<<<gN, blk256, 0, stream>>>(bufA, bufH, NTOT);
    // 5) u = m1 GEMM -> f32
    k_gemm_bf16<0><<<gGemm192, blk128, 0, stream>>>(wfM1, m1_b, bufH, bufA, nullptr, CC, CC);
    // 6) gelu(inorm(u)) -> bf16
    k_stat_part  <<<gStat, blk192, 0, stream>>>(bufA, part);
    k_stat_reduce<<<gRed,  blk256, 0, stream>>>(part, stats);
    k_norm_gelu  <<<gN,    blk256, 0, stream>>>(bufA, stats, bufH, NTOT);
    // 7) x_mod = x * sigmoid(m2 GEMM) -> bf16
    k_gemm_bf16<2><<<gGemm192, blk128, 0, stream>>>(wfM2, m2_b, bufH, bufX, x, CC, CC);
    // 8) qkv GEMM -> bf16 (axis-invariant; computed once)
    k_gemm_bf16<1><<<gGemmQKV, blk128, 0, stream>>>(wfQkv, qkv_b, bufX, bufQ, nullptr, QKV_M, CC);
    // 9) attention accumulator = 0
    k_zero_f32<<<gN, blk256, 0, stream>>>(bufA, NTOT);
    // 10) three axial attentions (sequential launches -> race-free +=)
    k_attn_axis<<<gAttn, blk32, 0, stream>>>(bufQ, A_d, 0, 2, bufA);
    k_attn_axis<<<gAttn, blk32, 0, stream>>>(bufQ, A_h, 1, 1, bufA);
    k_attn_axis<<<gAttn, blk32, 0, stream>>>(bufQ, A_w, 2, 0, bufA);
    // 11) attention sum -> bf16
    k_cvt_f32_bf16<<<gN, blk256, 0, stream>>>(bufA, bufH, NTOT);
    // 12) final projection -> d_out (channel-major f32)
    k_gemm_bf16<3><<<gGemm192, blk128, 0, stream>>>(wfProj, proj_b, bufH, (float*)d_out, nullptr, CC, CC);

    (void)in_sizes; (void)n_in; (void)out_size; (void)ws_size;
}
